// ForwardEulerSimulator_21663815041359
// MI455X (gfx1250) — matrix operand order repaired and checked by reading, not compiled, not run on hardware
//
#include <hip/hip_runtime.h>

typedef __attribute__((ext_vector_type(16))) _Float16 v16h;
typedef __attribute__((ext_vector_type(8)))  _Float16 v8h;
typedef __attribute__((ext_vector_type(8)))  float    v8f;
typedef __attribute__((ext_vector_type(4)))  float    v4f;

#define Q      2048
#define NX     32
#define NU     16
#define MSTEPS 512
#define HID    256

// LDS strides in elements (16B-aligned rows, bank-conflict-free)
#define W1T_S 72    // w1t[n][72] f16  : k=0..47 real, 48..63 zero, pad
#define W2T_S 264   // w2t[n][264] f16 : k=0..255, pad
#define XL_S  36    // x_lds[m][36] f32: n=0..31 real, pad
#define HS_S  72    // h_stage[w][m][72] f16: k_local=0..63 real, pad

// Hardware tanh on the CDNA5 TRANS32 pipe (v_tanh_f32); branch-free fallback.
__device__ __forceinline__ float fast_tanh(float x) {
#if __has_builtin(__builtin_amdgcn_tanhf)
  return __builtin_amdgcn_tanhf(x);
#else
  float e = __builtin_amdgcn_exp2f(x * 2.8853900817779268f);
  return 1.0f - 2.0f * __builtin_amdgcn_rcpf(e + 1.0f);
#endif
}

__device__ __forceinline__ v16h load_bfrag(const _Float16* p) {
  v8h lo = *(const v8h*)p;
  v8h hi = *(const v8h*)(p + 16);
  return __builtin_shufflevector(lo, hi, 0,1,2,3,4,5,6,7,8,9,10,11,12,13,14,15);
}

__global__ __launch_bounds__(128)
void euler_mlp_scan_kernel(const float* __restrict__ x0,
                           const float* __restrict__ u,
                           const float* __restrict__ W1,
                           const float* __restrict__ b1,
                           const float* __restrict__ W2,
                           const float* __restrict__ b2,
                           float* __restrict__ out) {
  __shared__ _Float16 w1t[HID][W1T_S];       // 36,864 B  (W1^T, f16)
  __shared__ _Float16 w2t[NX][W2T_S];        // 16,896 B  (W2^T, f16)
  __shared__ float    x_lds[16][XL_S];       //  2,304 B  (state x, f32)
  __shared__ _Float16 h_stage[4][16][HS_S];  //  9,216 B  (per-wave h, f16)

  const int tid    = threadIdx.x;
  const int wave   = tid >> 5;
  const int ln     = tid & 31;
  const int lm     = ln & 15;        // row (A) or column (B/C) sub-index
  const int hi     = ln >> 4;        // lane half select
  const int base8  = hi << 3;
  const int base16 = hi << 4;
  const int qbase  = blockIdx.x << 4;

  // ---- one-time init: weights -> LDS (transposed, f16), x0 -> LDS ----
  for (int idx = tid; idx < 48 * HID; idx += 128) {       // W1 is [k][n], 48x256
    int k = idx >> 8, n = idx & 255;
    w1t[n][k] = (_Float16)W1[idx];
  }
  for (int idx = tid; idx < HID * 16; idx += 128) {       // zero k = 48..63
    int n = idx >> 4, k = 48 + (idx & 15);
    w1t[n][k] = (_Float16)0.f;
  }
  for (int idx = tid; idx < HID * NX; idx += 128) {       // W2 is [k][n], 256x32
    int k = idx >> 5, n = idx & 31;
    w2t[n][k] = (_Float16)W2[idx];
  }
  for (int idx = tid; idx < 16 * NX; idx += 128) {        // x0 tile
    int mm = idx >> 5, n = idx & 31;
    x_lds[mm][n] = x0[(size_t)(qbase + mm) * NX + n];
  }
  const float b2v = b2[ln];

  __syncthreads();

  // ---- hoist loop-invariant weight fragments + bias vectors into VGPRs ----
  v16h w1A[4], w1B[4];     // GEMM1 B frags: 4 hidden N-tiles x 2 K-chunks
  v16h w2F[2][2];          // GEMM2 B frags: 2 K-chunks x 2 output N-tiles
  v8f  b1vec[4];           // bias as ready-made WMMA C operands
#pragma unroll
  for (int i = 0; i < 4; ++i) {
    const int n = ((wave << 2) + i) * 16 + lm;            // hidden column
    w1A[i] = load_bfrag(&w1t[n][base8]);                  // K = 0..31  (x part)
    w1B[i] = load_bfrag(&w1t[n][32 + base8]);             // K = 32..63 (u part)
    const float bv = b1[((wave << 2) + i) * 16 + lm];
#pragma unroll
    for (int r = 0; r < 8; ++r) b1vec[i][r] = bv;
  }
#pragma unroll
  for (int c = 0; c < 2; ++c) {
    const int kg = (wave << 6) + (c << 5) + base8;        // global K offset
    w2F[c][0] = load_bfrag(&w2t[lm][kg]);
    w2F[c][1] = load_bfrag(&w2t[16 + lm][kg]);
  }

  for (int t = 0; t < MSTEPS; ++t) {
    // ================= Phase A: emit x_t, build A fragments =================
    {
      // coalesced store of the 16x32 state tile (pre-update, matches scan)
      int e  = tid << 2;
      int mm = e >> 5, n = e & 31;
      v4f xv = *(const v4f*)&x_lds[mm][n];
      *(v4f*)(out + ((size_t)t * Q + qbase + mm) * NX + n) = xv;
    }
    v16h a0, a1;
    {
      // A fragment, K-chunk 0 (x columns 0..31), row M = lm
      const float* r = &x_lds[lm][0];
      v4f x0v = *(const v4f*)(r + base8);
      v4f x1v = *(const v4f*)(r + base8 + 4);
      v4f x2v = *(const v4f*)(r + base8 + 16);
      v4f x3v = *(const v4f*)(r + base8 + 20);
#pragma unroll
      for (int j = 0; j < 4; ++j) {
        a0[j]      = (_Float16)x0v[j];
        a0[j + 4]  = (_Float16)x1v[j];
        a0[j + 8]  = (_Float16)x2v[j];
        a0[j + 12] = (_Float16)x3v[j];
      }
    }
    {
      // A fragment, K-chunk 1 (u columns 32..47, zeros 48..63)
      const float* up = u + ((size_t)t * Q + qbase + lm) * NU + base8;
      v4f u0 = *(const v4f*)(up);
      v4f u1 = *(const v4f*)(up + 4);
#pragma unroll
      for (int j = 0; j < 4; ++j) {
        a1[j]      = (_Float16)u0[j];
        a1[j + 4]  = (_Float16)u1[j];
        a1[j + 8]  = (_Float16)0.f;
        a1[j + 12] = (_Float16)0.f;
      }
    }
    __syncthreads();   // all reads of x_lds done before dx accumulation

    // ================= Phase B: GEMM1 + tanh + GEMM2 partial ================
#pragma unroll
    for (int i = 0; i < 4; ++i) {
      v8f acc = __builtin_amdgcn_wmma_f32_16x16x32_f16(false, a0, false, w1A[i],
                                                       (short)0, b1vec[i],
                                                       false, false);
      acc = __builtin_amdgcn_wmma_f32_16x16x32_f16(false, a1, false, w1B[i],
                                                   (short)0, acc, false, false);
#pragma unroll
      for (int r = 0; r < 8; ++r) {
        float hv = fast_tanh(acc[r]);
        h_stage[wave][r + base8][(i << 4) + lm] = (_Float16)hv;  // row-major h
      }
    }

    // GEMM2 over this wave's K-slice (global K = 64*wave .. +63), C starts at 0
    v8f dx0 = {};
    v8f dx1 = {};
#pragma unroll
    for (int c = 0; c < 2; ++c) {
      v8h l0 = *(const v8h*)&h_stage[wave][lm][(c << 5) + base8];
      v8h l1 = *(const v8h*)&h_stage[wave][lm][(c << 5) + 16 + base8];
      v16h ha = __builtin_shufflevector(l0, l1, 0,1,2,3,4,5,6,7,8,9,10,11,12,13,14,15);
      dx0 = __builtin_amdgcn_wmma_f32_16x16x32_f16(false, ha, false, w2F[c][0],
                                                   (short)0, dx0, false, false);
      dx1 = __builtin_amdgcn_wmma_f32_16x16x32_f16(false, ha, false, w2F[c][1],
                                                   (short)0, dx1, false, false);
    }

    // reduce partial dx into the LDS-resident state: x += dx  (ds_add_f32)
#pragma unroll
    for (int r = 0; r < 8; ++r) {
      atomicAdd(&x_lds[r + base8][lm],      dx0[r]);
      atomicAdd(&x_lds[r + base8][16 + lm], dx1[r]);
    }
    if (wave == 3) {  // fold the b2 bias into the state update
#pragma unroll
      for (int mm = 0; mm < 16; ++mm) atomicAdd(&x_lds[mm][ln], b2v);
    }
    __syncthreads();  // x_{t+1} finalized for next iteration
  }
}

extern "C" void kernel_launch(void* const* d_in, const int* in_sizes, int n_in,
                              void* d_out, int out_size, void* d_ws, size_t ws_size,
                              hipStream_t stream) {
  (void)in_sizes; (void)n_in; (void)out_size; (void)d_ws; (void)ws_size;
  const float* x0 = (const float*)d_in[0];
  const float* u  = (const float*)d_in[1];
  const float* W1 = (const float*)d_in[2];
  const float* b1 = (const float*)d_in[3];
  const float* W2 = (const float*)d_in[4];
  const float* b2 = (const float*)d_in[5];
  float* out = (float*)d_out;

  dim3 grid(Q / 16);   // 128 workgroups, one 16-row batch tile each
  dim3 block(128);     // 4 wave32's cooperating per tile
  euler_mlp_scan_kernel<<<grid, block, 0, stream>>>(x0, u, W1, b1, W2, b2, out);
}
